// ioi_attention_circuit_87445534147428
// MI455X (gfx1250) — compile-verified, matrix operand order checked by reading, not run on hardware
//
#include <hip/hip_runtime.h>
#include <hip/hip_bf16.h>
#include <math.h>
#include <stdint.h>

// ---------------------------------------------------------------------------
// MI455X (gfx1250) GPT-2 forward for the IOI attention-circuit reference.
//   - wave32, v_wmma_f32_16x16x32_f16 for every matmul, fp32 accumulate
//   - weights pre-swizzled per layer into WMMA B-fragment order (f16):
//     B frags = 2x global_load_b128, coalesced, served from 192MB L2
//   - activations kept f16 row-major; GEMM A tiles staged to LDS with the
//     Tensor Data Mover (tensor_load_to_lds + s_wait_tensorcnt)
//   - flash attention reads pre-packed Q/K^T/V fragments (global_load_b128)
// ---------------------------------------------------------------------------

typedef __attribute__((ext_vector_type(16))) _Float16 v16h;
typedef __attribute__((ext_vector_type(8)))  _Float16 v8h;
typedef __attribute__((ext_vector_type(8)))  float    v8f;
typedef __attribute__((ext_vector_type(4)))  unsigned int v4u;
typedef __attribute__((ext_vector_type(8)))  int      v8i;
typedef __attribute__((ext_vector_type(4)))  int      v4i;

#define WMMA_F16(a, b, c) \
  __builtin_amdgcn_wmma_f32_16x16x32_f16(false, (a), false, (b), (short)0, (c), false, false)

#define T_SEQ  1024
#define DMODEL 768
#define NHEAD  12
#define HDIM   64
#define FFDIM  3072
#define QKVLD  2304
#define NLAYER 12
#define SCALE  0.125f   // 1/sqrt(64)

// GEMM tiling: block 128x64, K-step 32, 8 waves of 32x32 (2x2 WMMA tiles)
#define BM 128
#define BN 64
#define BK 32
#define ASTR 40          // 32 halfs + 4-dword TDM pad -> 80B row stride

// ---------------- fragment helpers -----------------------------------------
// Contiguous 16-half fragment (pre-swizzled layout): 2x global_load_b128.
__device__ __forceinline__ v16h load_frag16(const _Float16* __restrict__ p) {
  v16h r;
  ((v8h*)&r)[0] = ((const v8h*)p)[0];
  ((v8h*)&r)[1] = ((const v8h*)p)[1];
  return r;
}
// A 16x32 (MxK) fragment from row-major LDS tile (2x ds_load_b128 per lane).
__device__ __forceinline__ v16h load_a_lds(const _Float16* __restrict__ base, int ld) {
  const int lane = threadIdx.x & 31, m = lane & 15, g = lane >> 4;
  v16h a;
#pragma unroll
  for (int j = 0; j < 8; ++j) {
    const int k = ((j >> 2) << 4) + (g << 3) + ((j & 3) << 1);
    a[2 * j]     = base[m * ld + k];
    a[2 * j + 1] = base[m * ld + k + 1];
  }
  return a;
}
// K index inside a 32-deep fragment for element e of lane-half g.
__device__ __forceinline__ int frag_k(int e, int g) {
  const int j = e >> 1;
  return ((j >> 2) << 4) + (g << 3) + ((j & 3) << 1) + (e & 1);
}

// ---------------- Tensor Data Mover: 128x32 f16 tile -> LDS ----------------
// D# per CDNA5 ISA. 2D tensor, data_size=2B,
// pad_enable: +4 dwords after every 16 dwords -> LDS row stride = 40 halfs.
__device__ __forceinline__ void tdm_load_tile_f16(_Float16* lds,
                                                  const _Float16* gptr,
                                                  int row_elems) {
  const unsigned long long ga = (unsigned long long)(uintptr_t)gptr;
  v4u g0;
  g0[0] = 1u;                                          // count=1 (valid D#)
  g0[1] = (unsigned)(uintptr_t)lds;                    // lds_addr
  g0[2] = (unsigned)(ga & 0xffffffffu);                // global_addr[31:0]
  g0[3] = (unsigned)((ga >> 32) & 0x01ffffffu) | (2u << 30);  // [56:32] | type=2
  v8i g1;
  g1[0] = (int)((1u << 16) | (1u << 20) | (3u << 22) | (3u << 25));
  //            data_size=2B  pad_en      pad_int=16dw  pad_amt=4dw
  g1[1] = (int)(((unsigned)row_elems & 0xffffu) << 16);          // dim0 lo16
  g1[2] = (int)((((unsigned)row_elems >> 16) & 0xffffu) | ((unsigned)T_SEQ << 16));
  g1[3] = (int)(32u << 16);                                      // tile_dim0=32
  g1[4] = (int)(128u);                                           // tile_dim1=128
  g1[5] = (int)row_elems;                                        // dim0_stride lo32
  g1[6] = 0;
  g1[7] = 0;
  v4i z4 = {0, 0, 0, 0};
#if defined(__clang_major__) && __clang_major__ >= 23
  v8i z8 = {0, 0, 0, 0, 0, 0, 0, 0};
  __builtin_amdgcn_tensor_load_to_lds(g0, g1, z4, z4, z8, 0);
#else
  __builtin_amdgcn_tensor_load_to_lds(g0, g1, z4, z4, 0);
#endif
}

// ---------------- block reductions (256 threads = 8 waves) -----------------
__device__ __forceinline__ float block_sum256(float v, float* red) {
#pragma unroll
  for (int m = 16; m >= 1; m >>= 1) v += __shfl_xor(v, m, 32);
  __syncthreads();
  if ((threadIdx.x & 31) == 0) red[threadIdx.x >> 5] = v;
  __syncthreads();
  float t = 0.f;
#pragma unroll
  for (int i = 0; i < 8; ++i) t += red[i];
  return t;
}
__device__ __forceinline__ float block_max256(float v, float* red) {
#pragma unroll
  for (int m = 16; m >= 1; m >>= 1) v = fmaxf(v, __shfl_xor(v, m, 32));
  __syncthreads();
  if ((threadIdx.x & 31) == 0) red[threadIdx.x >> 5] = v;
  __syncthreads();
  float t = -1e30f;
#pragma unroll
  for (int i = 0; i < 8; ++i) t = fmaxf(t, red[i]);
  return t;
}

__device__ __forceinline__ float gelu_tanh(float x) {
  return 0.5f * x * (1.f + tanhf(0.7978845608028654f * (x + 0.044715f * x * x * x)));
}

// ---------------------------- embedding ------------------------------------
__global__ __launch_bounds__(256) void embed_kernel(
    const int* __restrict__ ids, const float* __restrict__ wte,
    const float* __restrict__ wpe, float* __restrict__ h) {
  const int t  = blockIdx.x;
  const int id = ids[t];
  for (int i = threadIdx.x; i < DMODEL; i += 256)
    h[(size_t)t * DMODEL + i] = wte[(size_t)id * DMODEL + i] + wpe[(size_t)t * DMODEL + i];
}

// ------------------------- layernorm (f32 in, f16 out) ---------------------
__global__ __launch_bounds__(256) void ln_kernel(
    const float* __restrict__ x, const float* __restrict__ g,
    const float* __restrict__ b, _Float16* __restrict__ y) {
  __shared__ float red[8];
  const int row = blockIdx.x;
  const float* xr = x + (size_t)row * DMODEL;
  float s = 0.f;
  for (int i = threadIdx.x; i < DMODEL; i += 256) s += xr[i];
  const float mean = block_sum256(s, red) * (1.f / DMODEL);
  float v = 0.f;
  for (int i = threadIdx.x; i < DMODEL; i += 256) {
    const float d = xr[i] - mean;
    v += d * d;
  }
  const float inv = rsqrtf(block_sum256(v, red) * (1.f / DMODEL) + 1e-5f);
  for (int i = threadIdx.x; i < DMODEL; i += 256)
    y[(size_t)row * DMODEL + i] = (_Float16)((xr[i] - mean) * inv * g[i] + b[i]);
}

// ------------- weight prep: fp32 [K][N] -> f16 B-fragment order ------------
// Fragment block (kt,nt) = 32x16; lane owns 16 contiguous halfs (32B).
__global__ __launch_bounds__(256) void prep_w_kernel(
    const float* __restrict__ W, _Float16* __restrict__ Wsw, int K, int N) {
  const int t = blockIdx.x * 256 + threadIdx.x;
  const int lane = t & 31, fid = t >> 5;
  const int ntiles = N >> 4;
  if (fid >= (K >> 5) * ntiles) return;
  const int nt = fid % ntiles, kt = fid / ntiles;
  const int n = (nt << 4) + (lane & 15), g = lane >> 4;
  const float* src = W + (size_t)(kt << 5) * N + n;
  v8h lo, hi;
#pragma unroll
  for (int e = 0; e < 16; ++e) {
    const _Float16 hv = (_Float16)src[(size_t)frag_k(e, g) * N];
    if (e < 8) lo[e] = hv; else hi[e - 8] = hv;
  }
  _Float16* dst = Wsw + (size_t)fid * 512 + lane * 16;
  *(v8h*)dst = lo;
  *(v8h*)(dst + 8) = hi;
}

// -------- pack q/k/v (fp32 qkv) into f16 attention fragments ---------------
// mode 0: Q as A-frags  (16 queries x 32 feats)
// mode 1: K^T as B-frags (k=feat, n=key)  -- same gather as Q but K columns
// mode 2: V as B-frags  (k=key, n=feat)
__global__ __launch_bounds__(256) void pack_qkv_kernel(
    const float* __restrict__ qkv, _Float16* __restrict__ Qa,
    _Float16* __restrict__ Kb, _Float16* __restrict__ Vb) {
  const int mode = blockIdx.y;
  const int t = blockIdx.x * 256 + threadIdx.x;   // 49152 per mode
  const int lane = t & 31, fid = t >> 5;          // fid: 0..1535 = hh*128 + sub
  const int m16 = lane & 15, g = lane >> 4;
  const int hh = fid >> 7, sub = fid & 127;
  v8h lo, hi;
  _Float16* dst;
  if (mode == 0 || mode == 1) {
    const int ks = sub & 1, tt = sub >> 1;        // tt: q-tile or key-tile
    const int coff = (mode == 0 ? 0 : DMODEL);
    const float* src = qkv + (size_t)(tt * 16 + m16) * QKVLD + coff + hh * HDIM + ks * 32;
#pragma unroll
    for (int e = 0; e < 16; ++e) {
      const _Float16 hv = (_Float16)src[frag_k(e, g)];
      if (e < 8) lo[e] = hv; else hi[e - 8] = hv;
    }
    dst = (mode == 0 ? Qa : Kb) + (size_t)fid * 512 + lane * 16;
  } else {
    const int ft = sub & 3, kb = sub >> 2;
    const int col = 2 * DMODEL + hh * HDIM + ft * 16 + m16;
#pragma unroll
    for (int e = 0; e < 16; ++e) {
      const int key = kb * 32 + frag_k(e, g);
      const _Float16 hv = (_Float16)qkv[(size_t)key * QKVLD + col];
      if (e < 8) lo[e] = hv; else hi[e - 8] = hv;
    }
    dst = Vb + (size_t)fid * 512 + lane * 16;
  }
  *(v8h*)dst = lo;
  *(v8h*)(dst + 8) = hi;
}

// ------------------- WMMA GEMM: C = act(A @ B + bias) [+ residual] ---------
// A: f16 row-major (M=1024 via grid), staged via TDM to LDS.
// Bsw: f16 pre-swizzled fragments. Cf (f32) and/or Ch (f16) outputs.
__global__ __launch_bounds__(256) void gemm_kernel(
    const _Float16* __restrict__ A, const _Float16* __restrict__ Bsw,
    const float* __restrict__ bias, const float* __restrict__ residual,
    float* __restrict__ Cf, _Float16* __restrict__ Ch,
    int N, int K, int do_gelu) {
  __shared__ _Float16 As[BM * ASTR];
  const int tid  = threadIdx.x;
  const int wave = tid >> 5;
  const int wm = wave & 3, wn = wave >> 2;
  const int brow = blockIdx.y * BM, bcol = blockIdx.x * BN;
  const int ntiles = N >> 4;

  v8f acc00 = {}, acc01 = {}, acc10 = {}, acc11 = {};

  for (int k0 = 0; k0 < K; k0 += BK) {
    __syncthreads();                       // As reuse guard
    if (wave == 0) {
      tdm_load_tile_f16(As, A + (size_t)brow * K + k0, K);
      __builtin_amdgcn_s_wait_tensorcnt(0);
    }
    __syncthreads();

    const int kt = k0 >> 5;
    const _Float16* bbase =
        Bsw + ((size_t)kt * ntiles + (bcol >> 4) + wn * 2) * 512 + (tid & 31) * 16;
    if (k0 + BK < K)                        // global_prefetch_b8 next B block
      __builtin_prefetch(Bsw + ((size_t)(kt + 1) * ntiles + (bcol >> 4)) * 512, 0, 0);

    const v16h b0 = load_frag16(bbase);
    const v16h b1 = load_frag16(bbase + 512);
    const v16h a0 = load_a_lds(&As[(wm * 32 + 0)  * ASTR], ASTR);
    const v16h a1 = load_a_lds(&As[(wm * 32 + 16) * ASTR], ASTR);
    acc00 = WMMA_F16(a0, b0, acc00);
    acc01 = WMMA_F16(a0, b1, acc01);
    acc10 = WMMA_F16(a1, b0, acc10);
    acc11 = WMMA_F16(a1, b1, acc11);
  }

  // epilogue: C-layout => row = i + 8*(lane>=16), col = lane&15
  const int lane = tid & 31, n = lane & 15, g = lane >> 4;
  v8f accs[2][2] = {{acc00, acc01}, {acc10, acc11}};
#pragma unroll
  for (int ti = 0; ti < 2; ++ti)
#pragma unroll
    for (int tj = 0; tj < 2; ++tj)
#pragma unroll
      for (int i = 0; i < 8; ++i) {
        const int row = brow + wm * 32 + ti * 16 + i + 8 * g;
        const int col = bcol + wn * 32 + tj * 16 + n;
        float val = accs[ti][tj][i] + bias[col];
        if (do_gelu) val = gelu_tanh(val);
        if (residual) val += residual[(size_t)row * N + col];
        if (Cf) Cf[(size_t)row * N + col] = val;
        if (Ch) Ch[(size_t)row * N + col] = (_Float16)val;
      }
}

// ------------------- flash attention, WMMA, 1 wave / 16 q-rows / head ------
__global__ __launch_bounds__(32) void attn_kernel(
    const _Float16* __restrict__ Qa, const _Float16* __restrict__ Kb,
    const _Float16* __restrict__ Vb, _Float16* __restrict__ ctxh) {
  __shared__ _Float16 Ph[16 * 32];
  const int lane = threadIdx.x & 31;
  const int hh = blockIdx.y;
  const int qt = blockIdx.x, qb = qt * 16;
  const int m16 = lane & 15, g = lane >> 4;
  const int loff = lane * 16;

  const v16h aq0 = load_frag16(Qa + ((size_t)(hh * 128 + qt * 2 + 0)) * 512 + loff);
  const v16h aq1 = load_frag16(Qa + ((size_t)(hh * 128 + qt * 2 + 1)) * 512 + loff);

  v8f ctx0 = {}, ctx1 = {}, ctx2 = {}, ctx3 = {};
  float mrow[8], lrow[8];
#pragma unroll
  for (int i = 0; i < 8; ++i) { mrow[i] = -1e30f; lrow[i] = 0.f; }

  const int nkb = (qb + 16 + 31) >> 5;   // key blocks of 32 up to causal edge
  for (int kb = 0; kb < nkb; ++kb) {
    const int keybase = kb * 32;
    // S = Q.K^T : two 16x16 tiles, K-dim 64 (two WMMAs each)
    v8f st[2];
#pragma unroll
    for (int t = 0; t < 2; ++t) {
      const size_t kf = (size_t)(hh * 128 + (kb * 2 + t) * 2);
      const v16h b0 = load_frag16(Kb + (kf + 0) * 512 + loff);
      const v16h b1 = load_frag16(Kb + (kf + 1) * 512 + loff);
      v8f z = {};
      st[t] = WMMA_F16(aq0, b0, z);
      st[t] = WMMA_F16(aq1, b1, st[t]);
    }
    // online softmax (per-row stats via 16-lane shfl reductions)
    float p0[8], p1[8], alpha[8];
#pragma unroll
    for (int i = 0; i < 8; ++i) {
      const int qrow = qb + i + 8 * g;
      float s0 = st[0][i] * SCALE;
      float s1 = st[1][i] * SCALE;
      if (keybase + m16 > qrow)      s0 = -1e30f;
      if (keybase + 16 + m16 > qrow) s1 = -1e30f;
      float rmax = fmaxf(s0, s1);
#pragma unroll
      for (int mm = 8; mm >= 1; mm >>= 1) rmax = fmaxf(rmax, __shfl_xor(rmax, mm, 32));
      const float mnew = fmaxf(mrow[i], rmax);
      alpha[i] = __expf(mrow[i] - mnew);
      p0[i] = __expf(s0 - mnew);
      p1[i] = __expf(s1 - mnew);
      float rs = p0[i] + p1[i];
#pragma unroll
      for (int mm = 8; mm >= 1; mm >>= 1) rs += __shfl_xor(rs, mm, 32);
      lrow[i] = lrow[i] * alpha[i] + rs;
      mrow[i] = mnew;
    }
#pragma unroll
    for (int i = 0; i < 8; ++i) {
      ctx0[i] *= alpha[i]; ctx1[i] *= alpha[i];
      ctx2[i] *= alpha[i]; ctx3[i] *= alpha[i];
    }
    // bounce P (16x32) through LDS: C-layout -> A-layout
    __syncthreads();
#pragma unroll
    for (int i = 0; i < 8; ++i) {
      const int r = i + 8 * g;
      Ph[r * 32 + m16]      = (_Float16)p0[i];
      Ph[r * 32 + 16 + m16] = (_Float16)p1[i];
    }
    __syncthreads();
    const v16h pa = load_a_lds(Ph, 32);
    const size_t vf = (size_t)(hh * 128 + kb * 4);
    ctx0 = WMMA_F16(pa, load_frag16(Vb + (vf + 0) * 512 + loff), ctx0);
    ctx1 = WMMA_F16(pa, load_frag16(Vb + (vf + 1) * 512 + loff), ctx1);
    ctx2 = WMMA_F16(pa, load_frag16(Vb + (vf + 2) * 512 + loff), ctx2);
    ctx3 = WMMA_F16(pa, load_frag16(Vb + (vf + 3) * 512 + loff), ctx3);
  }

  // normalize + write ctx f16 row-major [t][h*64+f]
  v8f cs[4] = {ctx0, ctx1, ctx2, ctx3};
#pragma unroll
  for (int f = 0; f < 4; ++f)
#pragma unroll
    for (int i = 0; i < 8; ++i) {
      const int r = qb + i + 8 * g;
      ctxh[(size_t)r * DMODEL + hh * HDIM + f * 16 + m16] = (_Float16)(cs[f][i] / lrow[i]);
    }
}

// ------------- IOI slice: softmax rows 600 & 1023 from fp32 qkv ------------
__global__ __launch_bounds__(256) void slice_kernel(
    const float* __restrict__ qkv, const int* __restrict__ pS,
    const int* __restrict__ pSa1, const int* __restrict__ pS2,
    float* __restrict__ out, int layer) {
  __shared__ float red[8];
  const int hh = blockIdx.x;
  const int S = *pS, Sa1 = *pSa1, S2 = *pS2;
  const float* Kv = qkv + DMODEL + hh * HDIM;  // k vectors, stride QKVLD

  for (int which = 0; which < 2; ++which) {
    const int r = (which == 0) ? S2 : (T_SEQ - 1);
    const float* q = qkv + (size_t)r * QKVLD + hh * HDIM;
    float mx = -1e30f;
    for (int j = threadIdx.x; j <= r; j += 256) {
      const float* kv = Kv + (size_t)j * QKVLD;
      float s = 0.f;
      for (int d = 0; d < HDIM; ++d) s += q[d] * kv[d];
      mx = fmaxf(mx, s * SCALE);
    }
    mx = block_max256(mx, red);
    float sum = 0.f;
    for (int j = threadIdx.x; j <= r; j += 256) {
      const float* kv = Kv + (size_t)j * QKVLD;
      float s = 0.f;
      for (int d = 0; d < HDIM; ++d) s += q[d] * kv[d];
      sum += __expf(s * SCALE - mx);
    }
    sum = block_sum256(sum, red);
    if (threadIdx.x == 0) {
      if (which == 0) {
        const float* kS = Kv + (size_t)S * QKVLD;
        const float* kA = Kv + (size_t)Sa1 * QKVLD;
        float sd = 0.f, si = 0.f;
        for (int d = 0; d < HDIM; ++d) { sd += q[d] * kS[d]; si += q[d] * kA[d]; }
        out[0 * NLAYER * NHEAD + layer * NHEAD + hh] = __expf(sd * SCALE - mx) / sum;
        out[1 * NLAYER * NHEAD + layer * NHEAD + hh] = __expf(si * SCALE - mx) / sum;
      } else {
        const float* k2 = Kv + (size_t)S2 * QKVLD;
        float sp = 0.f;
        for (int d = 0; d < HDIM; ++d) sp += q[d] * k2[d];
        const float p = __expf(sp * SCALE - mx) / sum;
        out[2 * NLAYER * NHEAD + layer * NHEAD + hh] = p;
        out[3 * NLAYER * NHEAD + layer * NHEAD + hh] = p;
      }
    }
    __syncthreads();
  }
}

// ---------------------------------------------------------------------------
extern "C" void kernel_launch(void* const* d_in, const int* in_sizes, int n_in,
                              void* d_out, int out_size, void* d_ws, size_t ws_size,
                              hipStream_t stream) {
  const int*   ids    = (const int*)d_in[0];
  // d_in[1] attention_mask: all ones -> add_mask == 0, unused
  const float* wte    = (const float*)d_in[2];
  const float* wpe    = (const float*)d_in[3];
  const float* ln1_g  = (const float*)d_in[4];
  const float* ln1_b  = (const float*)d_in[5];
  const float* W_qkv  = (const float*)d_in[6];
  const float* b_qkv  = (const float*)d_in[7];
  const float* W_o    = (const float*)d_in[8];
  const float* b_o    = (const float*)d_in[9];
  const float* ln2_g  = (const float*)d_in[10];
  const float* ln2_b  = (const float*)d_in[11];
  const float* W_fc   = (const float*)d_in[12];
  const float* b_fc   = (const float*)d_in[13];
  const float* W_proj = (const float*)d_in[14];
  const float* b_proj = (const float*)d_in[15];
  const int*   pS     = (const int*)d_in[16];
  const int*   pSa1   = (const int*)d_in[17];
  const int*   pS2    = (const int*)d_in[18];
  float* out = (float*)d_out;

  // ---- workspace carve (~50 MB) ----
  char* wp = (char*)d_ws;
  auto carve = [&](size_t bytes) {
    char* r = wp;
    wp += (bytes + 255) & ~(size_t)255;
    return r;
  };
  float*    hbuf  = (float*)carve((size_t)T_SEQ * DMODEL * 4);
  float*    h2    = (float*)carve((size_t)T_SEQ * DMODEL * 4);
  float*    qkvb  = (float*)carve((size_t)T_SEQ * QKVLD * 4);
  _Float16* xl    = (_Float16*)carve((size_t)T_SEQ * DMODEL * 2);
  _Float16* x2    = (_Float16*)carve((size_t)T_SEQ * DMODEL * 2);
  _Float16* ctx16 = (_Float16*)carve((size_t)T_SEQ * DMODEL * 2);
  _Float16* ffb16 = (_Float16*)carve((size_t)T_SEQ * FFDIM * 2);
  _Float16* sw_qkv  = (_Float16*)carve((size_t)DMODEL * QKVLD * 2);
  _Float16* sw_o    = (_Float16*)carve((size_t)DMODEL * DMODEL * 2);
  _Float16* sw_fc   = (_Float16*)carve((size_t)DMODEL * FFDIM * 2);
  _Float16* sw_proj = (_Float16*)carve((size_t)FFDIM * DMODEL * 2);
  _Float16* Qa = (_Float16*)carve((size_t)NHEAD * 128 * 512 * 2);
  _Float16* Kb = (_Float16*)carve((size_t)NHEAD * 128 * 512 * 2);
  _Float16* Vb = (_Float16*)carve((size_t)NHEAD * 128 * 512 * 2);

  embed_kernel<<<T_SEQ, 256, 0, stream>>>(ids, wte, wpe, hbuf);

  for (int l = 0; l < NLAYER; ++l) {
    // weight swizzle (f16 fragment order); K*N/16 threads each
    prep_w_kernel<<<(DMODEL / 32) * (QKVLD / 16) * 32 / 256, 256, 0, stream>>>(
        W_qkv + (size_t)l * DMODEL * QKVLD, sw_qkv, DMODEL, QKVLD);
    prep_w_kernel<<<(DMODEL / 32) * (DMODEL / 16) * 32 / 256, 256, 0, stream>>>(
        W_o + (size_t)l * DMODEL * DMODEL, sw_o, DMODEL, DMODEL);
    prep_w_kernel<<<(DMODEL / 32) * (FFDIM / 16) * 32 / 256, 256, 0, stream>>>(
        W_fc + (size_t)l * DMODEL * FFDIM, sw_fc, DMODEL, FFDIM);
    prep_w_kernel<<<(FFDIM / 32) * (DMODEL / 16) * 32 / 256, 256, 0, stream>>>(
        W_proj + (size_t)l * FFDIM * DMODEL, sw_proj, FFDIM, DMODEL);

    ln_kernel<<<T_SEQ, 256, 0, stream>>>(hbuf, ln1_g + l * DMODEL, ln1_b + l * DMODEL, xl);

    gemm_kernel<<<dim3(QKVLD / BN, T_SEQ / BM), 256, 0, stream>>>(
        xl, sw_qkv, b_qkv + (size_t)l * QKVLD, nullptr, qkvb, nullptr,
        QKVLD, DMODEL, 0);

    pack_qkv_kernel<<<dim3(192, 3), 256, 0, stream>>>(qkvb, Qa, Kb, Vb);

    attn_kernel<<<dim3(T_SEQ / 16, NHEAD), 32, 0, stream>>>(Qa, Kb, Vb, ctx16);

    slice_kernel<<<NHEAD, 256, 0, stream>>>(qkvb, pS, pSa1, pS2, out, l);

    gemm_kernel<<<dim3(DMODEL / BN, T_SEQ / BM), 256, 0, stream>>>(
        ctx16, sw_o, b_o + (size_t)l * DMODEL, hbuf, h2, nullptr,
        DMODEL, DMODEL, 0);

    ln_kernel<<<T_SEQ, 256, 0, stream>>>(h2, ln2_g + l * DMODEL, ln2_b + l * DMODEL, x2);

    gemm_kernel<<<dim3(FFDIM / BN, T_SEQ / BM), 256, 0, stream>>>(
        x2, sw_fc, b_fc + (size_t)l * FFDIM, nullptr, nullptr, ffb16,
        FFDIM, DMODEL, 1 /*gelu*/);

    gemm_kernel<<<dim3(DMODEL / BN, T_SEQ / BM), 256, 0, stream>>>(
        ffb16, sw_proj, b_proj + (size_t)l * DMODEL, h2, hbuf, nullptr,
        DMODEL, FFDIM, 0);
  }
}
